// GAT_33938831573707
// MI455X (gfx1250) — compile-verified, matrix operand order checked by reading, not run on hardware
//
#include <hip/hip_runtime.h>
#include <hip/hip_bf16.h>
#include <math.h>

// ---------------------------------------------------------------------------
// GAT (2-layer) for MI455X / gfx1250, wave32.
//  * Dense GEMMs: v_wmma_f32_16x16x32_bf16 on pre-packed bf16 fragments
//    (fragment-major layout -> inner loop is pure b128 loads + WMMA).
//  * Edge softmax + aggregation: streaming passes, float4 gathers, native
//    global f32 atomics (the true roofline limiter: ~8 GB @ 23.3 TB/s).
// ---------------------------------------------------------------------------

typedef __attribute__((ext_vector_type(16))) __bf16 v16bf;
typedef __attribute__((ext_vector_type(8)))  float  v8f;

#define NEG_SLOPE 0.2f

// K index striping of the 16-bit WMMA A/B operands (ISA 7.12.2):
// element i (0..15) of a lane's v16bf, hi = lane>>4:
__device__ __forceinline__ int kmap(int i, int hi) {
    const int r = i >> 1, b = i & 1;
    return (r < 4) ? (2 * r + b + 8 * hi) : (16 + 2 * (r - 4) + b + 8 * hi);
}

// ----------------------------- utility ------------------------------------

__global__ void gat_fill_f32(float* __restrict__ p, float v, long long n) {
    long long i = (long long)blockIdx.x * blockDim.x + threadIdx.x;
    if (i < n) p[i] = v;
}

// ------------------- bf16 fragment packing (one-time pass) -----------------
// A[M,K] f32 row-major -> Apack fragments: [mtile][ktile][lane][16 bf16]
// (each fragment = 32 lanes * 32 B, so the GEMM reads one aligned 32 B blob
//  per lane -> coalesced global_load_b128 pairs).  Optional fused ELU.
__global__ void gat_pack_a_bf16(const float* __restrict__ A,
                                __bf16* __restrict__ P,
                                int M, int K, int apply_elu) {
    const int t = blockIdx.x * blockDim.x + threadIdx.x;
    const int lane = t & 31;
    const int rest = t >> 5;
    const int ktiles = K >> 5;
    const int kt = rest % ktiles;
    const int mt = rest / ktiles;
    if (mt * 16 >= M) return;
    const int hi = lane >> 4, l15 = lane & 15;
    const float* a = A + (long long)(mt * 16 + l15) * K + kt * 32;
    v16bf frag;
#pragma unroll
    for (int i = 0; i < 16; ++i) {
        float v = a[kmap(i, hi)];
        if (apply_elu) v = (v > 0.0f) ? v : (__expf(v) - 1.0f);
        frag[i] = (__bf16)v;
    }
    *(v16bf*)(P + (((long long)mt * ktiles + kt) * 32 + lane) * 16) = frag;
}

// W[K,Ncol] f32 row-major -> Bpack fragments: [ntile][ktile][lane][16 bf16]
__global__ void gat_pack_b_bf16(const float* __restrict__ W,
                                __bf16* __restrict__ P,
                                int K, int Ncol) {
    const int t = blockIdx.x * blockDim.x + threadIdx.x;
    const int lane = t & 31;
    const int rest = t >> 5;
    const int ktiles = K >> 5;
    const int kt = rest % ktiles;
    const int nt = rest / ktiles;
    if (nt * 16 >= Ncol) return;
    const int hi = lane >> 4, l15 = lane & 15;
    const int col = nt * 16 + l15;
    v16bf frag;
#pragma unroll
    for (int i = 0; i < 16; ++i) {
        const int k = kt * 32 + kmap(i, hi);
        frag[i] = (__bf16)W[(long long)k * Ncol + col];
    }
    *(v16bf*)(P + (((long long)nt * ktiles + kt) * 32 + lane) * 16) = frag;
}

// ---------------------- packed-fragment WMMA GEMM --------------------------
// C[M,Ncol] = A @ W, f32 out.  One wave computes a 32x64 tile: 2 M-tiles x
// 4 N-tiles = 8 accumulators; per k-step: 6 fragment loads, 8 WMMAs.
// Requires M % 32 == 0, Ncol % 64 == 0, K % 32 == 0.
__global__ void gat_wmma_gemm_packed(const __bf16* __restrict__ Apack,
                                     const __bf16* __restrict__ Bpack,
                                     float* __restrict__ C,
                                     int M, int K, int Ncol) {
    const int wave = (blockIdx.x * blockDim.x + threadIdx.x) >> 5;
    const int lane = threadIdx.x & 31;
    const int ktiles = K >> 5;
    const int ngroups = Ncol >> 6;          // groups of 4 N-tiles
    const int mg = wave / ngroups;          // 32-row M-group
    const int ng = wave % ngroups;
    if (mg * 32 >= M) return;

    const v16bf* Ap = (const v16bf*)Apack;
    const v16bf* Bp = (const v16bf*)Bpack;
    const int mt0 = mg * 2, mt1 = mg * 2 + 1;

    v8f acc[2][4] = {};
    for (int kt = 0; kt < ktiles; ++kt) {
        const v16bf a0 = Ap[((long long)mt0 * ktiles + kt) * 32 + lane];
        const v16bf a1 = Ap[((long long)mt1 * ktiles + kt) * 32 + lane];
#pragma unroll
        for (int t = 0; t < 4; ++t) {
            const v16bf b = Bp[((long long)(ng * 4 + t) * ktiles + kt) * 32 + lane];
            acc[0][t] = __builtin_amdgcn_wmma_f32_16x16x32_bf16(
                false, a0, false, b, (short)0, acc[0][t], false, false);
            acc[1][t] = __builtin_amdgcn_wmma_f32_16x16x32_bf16(
                false, a1, false, b, (short)0, acc[1][t], false, false);
        }
    }

    const int hi = lane >> 4, l15 = lane & 15;
#pragma unroll
    for (int i = 0; i < 2; ++i)
#pragma unroll
        for (int t = 0; t < 4; ++t)
#pragma unroll
            for (int j = 0; j < 8; ++j) {
                const long long row = (long long)(mg * 32 + i * 16 + j + 8 * hi);
                C[row * Ncol + (ng * 4 + t) * 16 + l15] = acc[i][t][j];
            }
}

// --------------------- per-node attention dot scores -----------------------
// One wave per (n,h): lane-striped float4 loads (coalesced; one 512 B row
// per wave for D=128), then shfl_xor tree reduction.
template <int H, int D>
__global__ void gat_node_scores(const float* __restrict__ feat,
                                const float* __restrict__ al,
                                const float* __restrict__ ar,
                                float* __restrict__ el,
                                float* __restrict__ er, int N) {
    const long long wave = ((long long)blockIdx.x * blockDim.x + threadIdx.x) >> 5;
    const int lane = threadIdx.x & 31;
    if (wave >= (long long)N * H) return;
    const int n = (int)(wave / H), h = (int)(wave % H);
    float sl = 0.0f, sr = 0.0f;
    if (lane * 4 < D) {
        const float4 f = *(const float4*)&feat[(long long)n * H * D + h * D + lane * 4];
        const float4 a = *(const float4*)&al[h * D + lane * 4];
        const float4 r = *(const float4*)&ar[h * D + lane * 4];
        sl = f.x * a.x + f.y * a.y + f.z * a.z + f.w * a.w;
        sr = f.x * r.x + f.y * r.y + f.z * r.z + f.w * r.w;
    }
#pragma unroll
    for (int off = 16; off > 0; off >>= 1) {
        sl += __shfl_xor(sl, off, 32);
        sr += __shfl_xor(sr, off, 32);
    }
    if (lane == 0) { el[wave] = sl; er[wave] = sr; }
}

// --------------------------- float atomic max ------------------------------
// Ordered-int-bits trick: positives via signed atomicMax, negatives via
// unsigned atomicMin -> native global atomics, correct under mixed-sign
// races with -inf init.
__device__ __forceinline__ void atomicMaxF(float* addr, float v) {
    if (v >= 0.0f)
        atomicMax((int*)addr, __float_as_int(v));
    else
        atomicMin((unsigned int*)addr, (unsigned int)__float_as_int(v));
}

// ---------------------- edge pass 1: score + seg max -----------------------
// One thread per edge; unrolled over H so el/er/esc traffic is b128 for H=4.
template <int H>
__global__ void gat_edge_score_max(const int* __restrict__ src,
                                   const int* __restrict__ dst,
                                   const float* __restrict__ el,
                                   const float* __restrict__ er,
                                   float* __restrict__ esc,
                                   float* __restrict__ mbuf, long long E) {
    const long long e = (long long)blockIdx.x * blockDim.x + threadIdx.x;
    if (e >= E) return;
    const long long s = src[e], d = dst[e];
#pragma unroll
    for (int h = 0; h < H; ++h) {
        float v = el[s * H + h] + er[d * H + h];
        v = (v > 0.0f) ? v : NEG_SLOPE * v;          // leaky relu
        esc[e * H + h] = v;
        atomicMaxF(&mbuf[d * H + h], v);
    }
}

// ---------------------- edge pass 2: exp + seg sum -------------------------
template <int H>
__global__ void gat_edge_exp_sum(const int* __restrict__ dst,
                                 float* __restrict__ esc,
                                 const float* __restrict__ mbuf,
                                 float* __restrict__ sbuf, long long E) {
    const long long e = (long long)blockIdx.x * blockDim.x + threadIdx.x;
    if (e >= E) return;
    const long long d = dst[e];
#pragma unroll
    for (int h = 0; h < H; ++h) {
        const float ex = __expf(esc[e * H + h] - mbuf[d * H + h]);
        esc[e * H + h] = ex;
        atomicAdd(&sbuf[d * H + h], ex);
    }
}

// ------------------- edge pass 3: normalize + aggregate --------------------
// EPW edges per wave; each edge served by 32/EPW lanes doing float4 (b128)
// gathers of feat[src] and 4-wide f32 atomic adds to out[dst].
template <int H, int D, int EPW>
__global__ void gat_edge_aggregate(const int* __restrict__ src,
                                   const int* __restrict__ dst,
                                   const float* __restrict__ esc,
                                   const float* __restrict__ sbuf,
                                   const float* __restrict__ feat,
                                   float* __restrict__ out, long long E) {
    const int HD = H * D;
    const int LPE = 32 / EPW;
    const long long wave = ((long long)blockIdx.x * blockDim.x + threadIdx.x) >> 5;
    const int lane = threadIdx.x & 31;
    const long long e = wave * EPW + (lane / LPE);
    const int cl = lane % LPE;
    if (e >= E) return;
    const long long s = src[e];
    const long long d = dst[e];
    float alpha[H];
#pragma unroll
    for (int h = 0; h < H; ++h)
        alpha[h] = esc[e * H + h] / sbuf[d * H + h];
    for (int c = cl; c * 4 < HD; c += LPE) {
        const int idx = c * 4;
        const float a = alpha[idx / D];
        const float4 f = *(const float4*)&feat[s * HD + idx];
        float* o = &out[d * HD + idx];
        atomicAdd(o + 0, a * f.x);
        atomicAdd(o + 1, a * f.y);
        atomicAdd(o + 2, a * f.z);
        atomicAdd(o + 3, a * f.w);
    }
}

// ---------------------------------------------------------------------------

static inline int blocks_for(long long threads, int bs) {
    return (int)((threads + bs - 1) / bs);
}

extern "C" void kernel_launch(void* const* d_in, const int* in_sizes, int n_in,
                              void* d_out, int out_size, void* d_ws, size_t ws_size,
                              hipStream_t stream) {
    const float* x   = (const float*)d_in[0];   // (N,128)
    const float* W1  = (const float*)d_in[1];   // (128,512)
    const float* al1 = (const float*)d_in[2];   // (4,128)
    const float* ar1 = (const float*)d_in[3];   // (4,128)
    const float* W2  = (const float*)d_in[4];   // (512,64)
    const float* al2 = (const float*)d_in[5];   // (1,64)
    const float* ar2 = (const float*)d_in[6];   // (1,64)
    const int*   src = (const int*)d_in[7];     // (E,)
    const int*   dst = (const int*)d_in[8];     // (E,)

    const int IN = 128, HID = 128, H1 = 4, OUT = 64;
    const int N = in_sizes[0] / IN;             // 100000 (= 32 * 3125)
    const long long E = in_sizes[7];            // 3200000
    const int F1 = H1 * HID;                    // 512

    // ---- workspace carve-up (float units; every region 32B aligned) ----
    float*  ws     = (float*)d_ws;
    float*  feat1  = ws;                                  // N*512
    float*  agg1   = feat1 + (long long)N * F1;           // N*512
    float*  esc    = agg1  + (long long)N * F1;           // E*4 (then E*1)
    float*  el     = esc   + E * H1;                      // N*4
    float*  er     = el    + (long long)N * H1;           // N*4
    float*  mbuf   = er    + (long long)N * H1;           // N*4
    float*  sbuf   = mbuf  + (long long)N * H1;           // N*4
    __bf16* xpack  = (__bf16*)(sbuf + (long long)N * H1); // N*128 bf16
    __bf16* hpack  = xpack + (long long)N * IN;           // N*512 bf16
    __bf16* w1pack = hpack + (long long)N * F1;           // 128*512 bf16
    __bf16* w2pack = w1pack + (long long)IN * F1;         // 512*64 bf16
    float*  feat2  = feat1;                               // N*64 reuse
    float*  outw   = (float*)d_out;                       // N*64

    const int BS = 256;
    const float NEG_INF = -__builtin_huge_valf();

    // ---- pack weights + x to bf16 fragments ----
    gat_pack_b_bf16<<<blocks_for((long long)(F1 / 16) * (IN / 32) * 32, BS), BS, 0, stream>>>(
        W1, w1pack, IN, F1);
    gat_pack_b_bf16<<<blocks_for((long long)(OUT / 16) * (F1 / 32) * 32, BS), BS, 0, stream>>>(
        W2, w2pack, F1, OUT);
    gat_pack_a_bf16<<<blocks_for((long long)(N / 16) * (IN / 32) * 32, BS), BS, 0, stream>>>(
        x, xpack, N, IN, 0);

    // ================= Layer 1 =================
    {   // feat1 = x @ W1 : waves = (N/32) * (512/64)
        long long waves = ((long long)N / 32) * (F1 / 64);
        gat_wmma_gemm_packed<<<blocks_for(waves * 32, BS), BS, 0, stream>>>(
            xpack, w1pack, feat1, N, IN, F1);
    }
    gat_node_scores<H1, HID><<<blocks_for((long long)N * H1 * 32, BS), BS, 0, stream>>>(
        feat1, al1, ar1, el, er, N);

    gat_fill_f32<<<blocks_for((long long)N * H1, BS), BS, 0, stream>>>(mbuf, NEG_INF, (long long)N * H1);
    gat_fill_f32<<<blocks_for((long long)N * H1, BS), BS, 0, stream>>>(sbuf, 0.0f, (long long)N * H1);
    gat_fill_f32<<<blocks_for((long long)N * F1, BS), BS, 0, stream>>>(agg1, 0.0f, (long long)N * F1);

    gat_edge_score_max<H1><<<blocks_for(E, BS), BS, 0, stream>>>(src, dst, el, er, esc, mbuf, E);
    gat_edge_exp_sum<H1><<<blocks_for(E, BS), BS, 0, stream>>>(dst, esc, mbuf, sbuf, E);
    gat_edge_aggregate<H1, HID, 1><<<blocks_for(E * 32, BS), BS, 0, stream>>>(
        src, dst, esc, sbuf, feat1, agg1, E);

    // ---- ELU fused into bf16 packing of h for GEMM2 ----
    gat_pack_a_bf16<<<blocks_for((long long)(N / 16) * (F1 / 32) * 32, BS), BS, 0, stream>>>(
        agg1, hpack, N, F1, 1);

    // ================= Layer 2 =================
    {   // feat2 = elu(h) @ W2 : waves = (N/32) * (64/64)
        long long waves = (long long)N / 32;
        gat_wmma_gemm_packed<<<blocks_for(waves * 32, BS), BS, 0, stream>>>(
            hpack, w2pack, feat2, N, F1, OUT);
    }
    gat_node_scores<1, OUT><<<blocks_for((long long)N * 32, BS), BS, 0, stream>>>(
        feat2, al2, ar2, el, er, N);

    gat_fill_f32<<<blocks_for((long long)N, BS), BS, 0, stream>>>(mbuf, NEG_INF, (long long)N);
    gat_fill_f32<<<blocks_for((long long)N, BS), BS, 0, stream>>>(sbuf, 0.0f, (long long)N);
    gat_fill_f32<<<blocks_for((long long)N * OUT, BS), BS, 0, stream>>>(outw, 0.0f, (long long)N * OUT);

    gat_edge_score_max<1><<<blocks_for(E, BS), BS, 0, stream>>>(src, dst, el, er, esc, mbuf, E);
    gat_edge_exp_sum<1><<<blocks_for(E, BS), BS, 0, stream>>>(dst, esc, mbuf, sbuf, E);
    gat_edge_aggregate<1, OUT, 2><<<blocks_for(((E + 1) / 2) * 32, BS), BS, 0, stream>>>(
        src, dst, esc, sbuf, feat2, outw, E);
}
